// RNN_44246753084122
// MI455X (gfx1250) — matrix operand order repaired and checked by reading, not compiled, not run on hardware
//
#include <hip/hip_runtime.h>
#include <hip/hip_bf16.h>
#include <math.h>

#define S_LEN 1024
#define I_DIM 1024
#define H_DIM 2048
#define L_NUM 4

typedef __attribute__((ext_vector_type(16))) _Float16 v16h;
typedef __attribute__((ext_vector_type(8)))  _Float16 v8h;
typedef __attribute__((ext_vector_type(8)))  float    v8f;

union V16H { v16h v; v8h h[2]; };

__device__ __forceinline__ v8f wmma16(v16h a, v16h b, v8f c) {
  // (neg_a, A, neg_b, B, c_mod, C, reuse_a, reuse_b) -> v_wmma_f32_16x16x32_f16
  return __builtin_amdgcn_wmma_f32_16x16x32_f16(false, a, false, b, (short)0, c,
                                                false, false);
}

// A operand, 16x32 f16 tile from row-major W[ld], rows [row0,row0+16), K [k0,k0+32)
// lane<16 : row=row0+lane,    K = k0+0..15 contiguous
// lane>=16: row=row0+lane-16, K = k0+16..31 contiguous
__device__ __forceinline__ v16h load_a16(const _Float16* W, int ld, int row0,
                                         int k0, int lane) {
  int l = lane & 15, hi = lane >> 4;
  const _Float16* p = W + (size_t)(row0 + l) * (size_t)ld + (size_t)(k0 + (hi << 4));
  V16H r;
  r.h[0] = *(const v8h*)p;
  r.h[1] = *(const v8h*)(p + 8);
  return r.v;
}

// B operand, 32x16 f16 tile with B[k][n] = W[col0+n][k0+k] (W row-major, ld stride)
// lane<16 : col=col0+lane,    K = k0+0..15 contiguous
// lane>=16: col=col0+lane-16, K = k0+16..31 contiguous
__device__ __forceinline__ v16h load_b16(const _Float16* W, int ld, int col0,
                                         int k0, int lane) {
  int l = lane & 15, hi = lane >> 4;
  const _Float16* p = W + (size_t)(col0 + l) * (size_t)ld + (size_t)(k0 + (hi << 4));
  V16H r;
  r.h[0] = *(const v8h*)p;
  r.h[1] = *(const v8h*)(p + 8);
  return r.v;
}

// Broadcast-B for GEMV: every column of B equals h[k0..k0+31] -> all C columns equal
__device__ __forceinline__ v16h load_bvec(const _Float16* hv, int k0, int lane) {
  int hi = lane >> 4;
  const _Float16* p = hv + k0 + (hi << 4);
  V16H r;
  r.h[0] = *(const v8h*)p;
  r.h[1] = *(const v8h*)(p + 8);
  return r.v;
}

__global__ __launch_bounds__(256) void cvt_f32_f16(const float* __restrict__ s,
                                                   _Float16* __restrict__ d, int n) {
  for (int i = blockIdx.x * blockDim.x + threadIdx.x; i < n;
       i += gridDim.x * blockDim.x)
    d[i] = (_Float16)s[i];
}

__global__ __launch_bounds__(256) void init_state(_Float16* h16, unsigned* bar) {
  int i = blockIdx.x * blockDim.x + threadIdx.x;
  if (i < L_NUM * H_DIM) h16[i] = (_Float16)0.0f;  // h(t=0) = 0 (ping buffer 0)
  if (i == 0) { bar[0] = 0u; bar[1] = 0u; }
}

// Z0[t][j] = sum_i X[t][i] * Wi0[j][i]   (full-efficiency WMMA GEMM, fp32 accum)
__global__ __launch_bounds__(256) void z0_gemm(const _Float16* __restrict__ hX,
                                               const _Float16* __restrict__ hWi0,
                                               float* __restrict__ Z0) {
  const int lane = threadIdx.x & 31;
  const int wid  = threadIdx.x >> 5;
  const int gw = blockIdx.x * 8 + wid;   // 0..8191 wave-tiles
  const int tT = gw >> 7;                // 64 tiles of t
  const int tJ = gw & 127;               // 128 tiles of j
  const int l15 = lane & 15, hi = lane >> 4;
  v8f acc = {};
#pragma unroll 4
  for (int k0 = 0; k0 < I_DIM; k0 += 32) {
    v16h a = load_a16(hX,   I_DIM, tT << 4, k0, lane);
    v16h b = load_b16(hWi0, I_DIM, tJ << 4, k0, lane);
    acc = wmma16(a, b, acc);
  }
#pragma unroll
  for (int r = 0; r < 8; ++r) {
    int row = (tT << 4) + (hi << 3) + r;   // C layout: VGPR r = rows r / r+8
    int col = (tJ << 4) + l15;
    Z0[(size_t)row * H_DIM + col] = acc[r];
  }
}

// Persistent recurrent kernel: 64 blocks x 8 waves. 128 groups (4 waves each)
// own 16 output rows; the 4 waves split K=2048 into 512-slices and reduce in LDS.
// Grid-wide sense-reversal barrier between layers (4 per step, 4096 total).
__global__ __launch_bounds__(256) void rnn_recur(
    const _Float16* __restrict__ hWiR, const _Float16* __restrict__ hWh,
    const float* __restrict__ Z0, const float* __restrict__ bi0,
    const float* __restrict__ biR, const float* __restrict__ bh,
    _Float16* __restrict__ h16, float* __restrict__ out, unsigned* bar) {
  const int lane  = threadIdx.x & 31;
  const int wid   = threadIdx.x >> 5;
  const int group = (int)(blockIdx.x << 1) | (wid >> 2);  // 0..127
  const int qw    = wid & 3;                              // K slice index
  const int row0  = group << 4;
  const int l15 = lane & 15, hi = lane >> 4;
  const int kbase = qw << 9;                              // qw * 512
  __shared__ float red[8][16];

#pragma unroll 1
  for (int t = 0; t < S_LEN; ++t) {
    const int p = t & 1;  // ping-pong: read prev from buf p, write cur to buf 1-p
    const _Float16* hprev = h16 + (size_t)p * (L_NUM * H_DIM);
    _Float16*       hcur  = h16 + (size_t)(1 - p) * (L_NUM * H_DIM);
#pragma unroll 1
    for (int l = 0; l < L_NUM; ++l) {
      v8f acc = {};
      {  // recurrent term: Wh[l] @ h_prev[l]  (broadcast-B WMMA chain)
        const _Float16* Wl = hWh + (size_t)l * H_DIM * H_DIM;
        const _Float16* hp = hprev + l * H_DIM;
#pragma unroll 4
        for (int kc = 0; kc < 16; ++kc) {
          int k0 = kbase + (kc << 5);
          acc = wmma16(load_a16(Wl, H_DIM, row0, k0, lane),
                       load_bvec(hp, k0, lane), acc);
        }
      }
      if (l > 0) {  // input term: Wi[l-1] @ h_cur[l-1]
        const _Float16* Wi = hWiR + (size_t)(l - 1) * H_DIM * H_DIM;
        const _Float16* hb = hcur + (l - 1) * H_DIM;
#pragma unroll 4
        for (int kc = 0; kc < 16; ++kc) {
          int k0 = kbase + (kc << 5);
          acc = wmma16(load_a16(Wi, H_DIM, row0, k0, lane),
                       load_bvec(hb, k0, lane), acc);
        }
      }
      // All C columns are equal; lanes 0 and 16 hold rows 0-7 / 8-15.
      if (l15 == 0) {
#pragma unroll
        for (int r = 0; r < 8; ++r) red[wid][(hi << 3) + r] = acc[r];
      }
      __syncthreads();
      if (qw == 0 && lane < 16) {  // reduce 4 K-slices, bias + tanh, publish
        int row = row0 + lane;
        float s = red[wid][lane] + red[wid + 1][lane] +
                  red[wid + 2][lane] + red[wid + 3][lane];
        if (l == 0) s += Z0[(size_t)t * H_DIM + row] + bi0[row];
        else        s += biR[(size_t)(l - 1) * H_DIM + row];
        s += bh[(size_t)l * H_DIM + row];
        s = tanhf(s);
        hcur[(size_t)l * H_DIM + row] = (_Float16)s;
        if (l == L_NUM - 1) out[(size_t)t * H_DIM + row] = s;
      }
      // ---- grid-wide barrier (release h16 stores, acquire for next layer) ----
      __threadfence();
      __syncthreads();
      if (threadIdx.x == 0) {
        unsigned sv = __hip_atomic_load(&bar[1], __ATOMIC_RELAXED,
                                        __HIP_MEMORY_SCOPE_AGENT);
        unsigned a = __hip_atomic_fetch_add(&bar[0], 1u, __ATOMIC_ACQ_REL,
                                            __HIP_MEMORY_SCOPE_AGENT);
        if (a == gridDim.x - 1u) {
          __hip_atomic_store(&bar[0], 0u, __ATOMIC_RELAXED,
                             __HIP_MEMORY_SCOPE_AGENT);
          __hip_atomic_fetch_add(&bar[1], 1u, __ATOMIC_RELEASE,
                                 __HIP_MEMORY_SCOPE_AGENT);
        } else {
          while (__hip_atomic_load(&bar[1], __ATOMIC_ACQUIRE,
                                   __HIP_MEMORY_SCOPE_AGENT) == sv)
            __builtin_amdgcn_s_sleep(1);
        }
      }
      __syncthreads();
      __threadfence();
    }
  }
}

extern "C" void kernel_launch(void* const* d_in, const int* in_sizes, int n_in,
                              void* d_out, int out_size, void* d_ws, size_t ws_size,
                              hipStream_t stream) {
  (void)in_sizes; (void)n_in; (void)out_size; (void)ws_size;
  const float* X   = (const float*)d_in[0];  // (1,S,I)
  const float* Wi0 = (const float*)d_in[1];  // (H,I)
  const float* bi0 = (const float*)d_in[2];  // (H)
  const float* WiR = (const float*)d_in[3];  // (L-1,H,H)
  const float* biR = (const float*)d_in[4];  // (L-1,H)
  const float* Wh  = (const float*)d_in[5];  // (L,H,H)
  const float* bh  = (const float*)d_in[6];  // (L,H)
  float* out = (float*)d_out;                // (S,1,H) f32

  char* ws = (char*)d_ws;
  const size_t MB = 1024 * 1024;
  _Float16* hX   = (_Float16*)(ws);                  //  2 MB : X f16
  _Float16* hWi0 = (_Float16*)(ws + 2 * MB);         //  4 MB : Wi0 f16
  _Float16* hWiR = (_Float16*)(ws + 6 * MB);         // 24 MB : Wi_rest f16
  _Float16* hWh  = (_Float16*)(ws + 30 * MB);        // 32 MB : Wh f16
  float*    Z0   = (float*)(ws + 62 * MB);           //  8 MB : X @ Wi0^T
  _Float16* h16  = (_Float16*)(ws + 70 * MB);        // 32 KB : hidden ping-pong
  unsigned* bar  = (unsigned*)(ws + 70 * MB + 65536);

  // 1) convert inputs/weights to f16 (weights ~60MB -> L2-resident)
  cvt_f32_f16<<<2048, 256, 0, stream>>>(X,   hX,   S_LEN * I_DIM);
  cvt_f32_f16<<<2048, 256, 0, stream>>>(Wi0, hWi0, H_DIM * I_DIM);
  cvt_f32_f16<<<2048, 256, 0, stream>>>(WiR, hWiR, (L_NUM - 1) * H_DIM * H_DIM);
  cvt_f32_f16<<<2048, 256, 0, stream>>>(Wh,  hWh,  L_NUM * H_DIM * H_DIM);
  // 2) zero initial hidden state + barrier
  init_state<<<32, 256, 0, stream>>>(h16, bar);
  // 3) hoisted layer-0 input GEMM via WMMA
  z0_gemm<<<1024, 256, 0, stream>>>(hX, hWi0, Z0);
  // 4) persistent recurrent kernel (all 1024 steps, 4 layers)
  rnn_recur<<<64, 256, 0, stream>>>(hWiR, hWh, Z0, bi0, biR, bh, h16, out, bar);
}
